// AttentionBlock_90890097918069
// MI455X (gfx1250) — matrix operand order repaired and checked by reading, not compiled, not run on hardware
//
#include <hip/hip_runtime.h>

typedef __bf16 bf16;
typedef __attribute__((ext_vector_type(16))) __bf16 v16bf;
typedef __attribute__((ext_vector_type(8)))  float  v8f;

#define Bc 2
#define Tc 2048
#define Dc 1024
#define NHEAD 16
#define Hc 64

struct Pair { uint4 a, b; };

// A-fragment (16x32 bf16): lane half hi reads contiguous K = base+{0..15}
__device__ __forceinline__ v16bf load_frag_a(const bf16* p) {
    Pair pr;
    pr.a = *reinterpret_cast<const uint4*>(p);
    pr.b = *reinterpret_cast<const uint4*>(p + 8);
    return __builtin_bit_cast(v16bf, pr);
}
// B-fragment (32x16 bf16): lane half hi reads contiguous K = base+{0..15}
__device__ __forceinline__ v16bf load_frag_b(const bf16* p) {
    Pair pr;
    pr.a = *reinterpret_cast<const uint4*>(p);
    pr.b = *reinterpret_cast<const uint4*>(p + 8);
    return __builtin_bit_cast(v16bf, pr);
}

__device__ __forceinline__ v8f wmma_bf16(v16bf a, v16bf b, v8f c) {
    return __builtin_amdgcn_wmma_f32_16x16x32_bf16(false, a, false, b, (short)0, c,
                                                   false, false);
}

// CDNA5 async global->LDS copy (16B per lane), tracked by ASYNCcnt.
__device__ __forceinline__ void async_load_b128(const bf16* gsrc, bf16* lds_dst) {
    const unsigned int  l = (unsigned int)(uintptr_t)lds_dst;       // LDS byte addr
    const unsigned long long g = (unsigned long long)(uintptr_t)gsrc;
    asm volatile("global_load_async_to_lds_b128 %0, %1, off"
                 :: "v"(l), "v"(g)
                 : "memory");
}
__device__ __forceinline__ void wait_async0() {
    asm volatile("s_wait_asynccnt 0" ::: "memory");
}

// ---------------------------------------------------------------------------
// GEMM: C[M x Nn] = A[M x K] * W[K x Nn] + bias. 128x128 block, BK=32, 8 waves.
// QKV epilogue scatters q/k/v (bf16 ws; k,v also f32 outputs).
// ---------------------------------------------------------------------------
template<bool A_BF16, bool QKV>
__global__ __launch_bounds__(256)
void gemm_bf16_wmma(const void* __restrict__ Aptr,
                    const float* __restrict__ W,
                    const float* __restrict__ bias,
                    float* __restrict__ Cout,
                    bf16* __restrict__ q_ws, bf16* __restrict__ k_ws,
                    bf16* __restrict__ v_ws,
                    float* __restrict__ k_out, float* __restrict__ v_out,
                    int K, int Nn)
{
    __shared__ __align__(16) bf16 Alds[128][40];   // [m][k], padded
    __shared__ __align__(16) bf16 Blds[128][40];   // [n][k], transposed, padded

    const int tid  = threadIdx.x;
    const int wid  = tid >> 5;
    const int lane = tid & 31;
    const int lr   = lane & 15;
    const int hi   = lane >> 4;
    const int wm   = wid >> 2;          // 0..1 -> 64 rows each
    const int wn   = wid & 3;           // 0..3 -> 32 cols each
    const int m0   = blockIdx.x * 128;
    const int n0   = blockIdx.y * 128;

    const v8f vzero = {0.f,0.f,0.f,0.f,0.f,0.f,0.f,0.f};
    v8f acc[4][2];
#pragma unroll
    for (int i = 0; i < 4; ++i)
#pragma unroll
        for (int j = 0; j < 2; ++j) acc[i][j] = vzero;

    for (int k0 = 0; k0 < K; k0 += 32) {
        if constexpr (A_BF16) {
            const bf16* A = (const bf16*)Aptr;
#pragma unroll
            for (int i = 0; i < 2; ++i) {
                int idx = tid + i * 256;              // 8-elem chunks, 512 total
                int r = idx >> 2, c8 = (idx & 3) << 3;
                uint4 u = *reinterpret_cast<const uint4*>(
                    A + (size_t)(m0 + r) * K + k0 + c8);
                *reinterpret_cast<uint4*>(&Alds[r][c8]) = u;
            }
        } else {
            const float* A = (const float*)Aptr;
#pragma unroll
            for (int i = 0; i < 4; ++i) {
                int idx = tid + i * 256;              // float4 chunks, 1024 total
                int r = idx >> 3, c4 = (idx & 7) << 2;
                float4 f = *reinterpret_cast<const float4*>(
                    A + (size_t)(m0 + r) * K + k0 + c4);
                Alds[r][c4 + 0] = (bf16)f.x;
                Alds[r][c4 + 1] = (bf16)f.y;
                Alds[r][c4 + 2] = (bf16)f.z;
                Alds[r][c4 + 3] = (bf16)f.w;
            }
        }
        // B tile 32x128, coalesced read along n, transposed store
#pragma unroll
        for (int i = 0; i < 16; ++i) {
            int idx = tid + i * 256;                  // 4096 elements
            int kk = idx >> 7, nn = idx & 127;
            Blds[nn][kk] = (bf16)W[(size_t)(k0 + kk) * Nn + n0 + nn];
        }
        __syncthreads();

        v16bf af[4], bfrag[2];
#pragma unroll
        for (int mt = 0; mt < 4; ++mt)
            af[mt] = load_frag_a(&Alds[wm * 64 + mt * 16 + lr][hi * 16]);
#pragma unroll
        for (int nt = 0; nt < 2; ++nt)
            bfrag[nt] = load_frag_b(&Blds[wn * 32 + nt * 16 + lr][hi * 16]);
#pragma unroll
        for (int mt = 0; mt < 4; ++mt)
#pragma unroll
            for (int nt = 0; nt < 2; ++nt)
                acc[mt][nt] = wmma_bf16(af[mt], bfrag[nt], acc[mt][nt]);
        __syncthreads();
    }

#pragma unroll
    for (int mt = 0; mt < 4; ++mt) {
#pragma unroll
        for (int nt = 0; nt < 2; ++nt) {
            const int col   = n0 + wn * 32 + nt * 16 + lr;
            const int rbase = m0 + wm * 64 + mt * 16 + hi * 8;
            const float bv  = bias[col];
#pragma unroll
            for (int e = 0; e < 8; ++e) {
                const int r = rbase + e;
                const float val = acc[mt][nt][e] + bv;
                if constexpr (QKV) {
                    if (col < Dc) {
                        q_ws[(size_t)r * Dc + col] = (bf16)val;
                    } else if (col < 2 * Dc) {
                        const int c = col - Dc;
                        k_out[(size_t)r * Dc + c] = val;
                        k_ws [(size_t)r * Dc + c] = (bf16)val;
                    } else {
                        const int c = col - 2 * Dc;
                        v_out[(size_t)r * Dc + c] = val;
                        v_ws [(size_t)r * Dc + c] = (bf16)val;
                    }
                } else {
                    Cout[(size_t)r * Nn + col] = val;
                }
            }
        }
    }
}

// ---------------------------------------------------------------------------
// Causal flash attention. One block per (batch, head, 128-row q block).
// 8 waves; each wave owns 16 q rows. KV blocks (64 wide) are double-buffered:
// async global->LDS DMA of block sb+1 overlaps the WMMA work of block sb.
// ---------------------------------------------------------------------------
__global__ __launch_bounds__(256)
void attn_flash_wmma(const bf16* __restrict__ q_ws,
                     const bf16* __restrict__ k_ws,
                     const bf16* __restrict__ v_ws,
                     bf16* __restrict__ attn_ws)
{
    __shared__ __align__(16) bf16 Klds[2][64][72];   // [buf][s][h] (padded)
    __shared__ __align__(16) bf16 Vraw[2][64][64];   // [buf][s][h] (async dest)
    __shared__ __align__(16) bf16 Vt  [64][72];      // [h][s] transposed
    __shared__ __align__(16) bf16 Plds[8][16][72];   // per-wave P transpose

    const int tid  = threadIdx.x;
    const int wid  = tid >> 5;
    const int lane = tid & 31;
    const int lr   = lane & 15;
    const int hi   = lane >> 4;
    const int bx   = blockIdx.x;
    const int qb   = bx & 15;            // q block within T (16 blocks of 128)
    const int head = (bx >> 4) & 15;
    const int b    = bx >> 8;
    const int qrow0 = qb * 128 + wid * 16;

    const v8f vzero = {0.f,0.f,0.f,0.f,0.f,0.f,0.f,0.f};
    const float NINF = -__builtin_inff();

    // Q fragments for this wave's 16 rows (H=64 -> two K=32 chunks)
    v16bf qfrag[2];
    {
        const bf16* qp = q_ws + ((size_t)(b * Tc + qrow0 + lr)) * Dc + head * Hc;
        qfrag[0] = load_frag_a(qp + hi * 16);
        qfrag[1] = load_frag_a(qp + 32 + hi * 16);
    }

    // async prefetch of one 64-row KV block (K and V), 16B chunks per lane
    auto prefetch = [&](int sb, int buf) {
#pragma unroll
        for (int i2 = 0; i2 < 2; ++i2) {
            int c = tid + i2 * 256;                  // 512 chunks of 8 bf16
            int i = c >> 3, h8 = (c & 7) << 3;
            const size_t gofs = ((size_t)(b * Tc + sb * 64 + i)) * Dc + head * Hc + h8;
            async_load_b128(k_ws + gofs, &Klds[buf][i][h8]);
            async_load_b128(v_ws + gofs, &Vraw[buf][i][h8]);
        }
    };

    v8f O[4];
#pragma unroll
    for (int n = 0; n < 4; ++n) O[n] = vzero;
    float mrow[8], lrow[8];
#pragma unroll
    for (int e = 0; e < 8; ++e) { mrow[e] = NINF; lrow[e] = 0.f; }

    const int nsb = 2 * qb + 2;          // causal KV block count
    prefetch(0, 0);
    for (int sb = 0; sb < nsb; ++sb) {
        const int cur = sb & 1;
        const int s0  = sb * 64;

        wait_async0();                   // own async transfers landed in LDS
        __syncthreads();                 // everyone's transfers visible
        if (sb + 1 < nsb) prefetch(sb + 1, cur ^ 1);   // overlap next block

        // LDS->LDS transpose of V into [h][s] B-fragment layout
#pragma unroll
        for (int i2 = 0; i2 < 16; ++i2) {
            int c = tid + i2 * 256;                  // 4096 elements
            int i = c >> 6, h = c & 63;
            Vt[h][i] = Vraw[cur][i][h];
        }
        __syncthreads();

        // S = Q * K^T for four 16-col tiles
        v8f S[4];
        bool valid[4];
#pragma unroll
        for (int j = 0; j < 4; ++j) {
            valid[j] = (s0 + j * 16) <= (qrow0 + 15);
            if (valid[j]) {
                v8f s = vzero;
                s = wmma_bf16(qfrag[0],
                              load_frag_b(&Klds[cur][j*16 + lr][hi*16]),      s);
                s = wmma_bf16(qfrag[1],
                              load_frag_b(&Klds[cur][j*16 + lr][32 + hi*16]), s);
                S[j] = s;
            }
        }
        // scale + causal mask + row max
        float rmax[8];
#pragma unroll
        for (int e = 0; e < 8; ++e) rmax[e] = NINF;
#pragma unroll
        for (int j = 0; j < 4; ++j) {
            if (!valid[j]) continue;
#pragma unroll
            for (int e = 0; e < 8; ++e) {
                const int qt  = qrow0 + hi * 8 + e;
                const int scc = s0 + j * 16 + lr;
                float v = (scc <= qt) ? S[j][e] * 0.125f : NINF;
                S[j][e] = v;
                rmax[e] = fmaxf(rmax[e], v);
            }
        }
#pragma unroll
        for (int e = 0; e < 8; ++e) {
            rmax[e] = fmaxf(rmax[e], __shfl_xor(rmax[e], 1, 32));
            rmax[e] = fmaxf(rmax[e], __shfl_xor(rmax[e], 2, 32));
            rmax[e] = fmaxf(rmax[e], __shfl_xor(rmax[e], 4, 32));
            rmax[e] = fmaxf(rmax[e], __shfl_xor(rmax[e], 8, 32));
        }
        float alpha[8];
#pragma unroll
        for (int e = 0; e < 8; ++e) {
            const float mo = mrow[e];
            const float mn = fmaxf(mo, rmax[e]);
            alpha[e] = (mn == mo) ? 1.0f : __expf(mo - mn);
            mrow[e]  = mn;
        }
        // P = exp(S - m), transpose through LDS, row sums
        float rsum[8];
#pragma unroll
        for (int e = 0; e < 8; ++e) rsum[e] = 0.f;
#pragma unroll
        for (int j = 0; j < 4; ++j) {
#pragma unroll
            for (int e = 0; e < 8; ++e) {
                float p = valid[j] ? __expf(S[j][e] - mrow[e]) : 0.f;
                rsum[e] += p;
                Plds[wid][hi * 8 + e][j * 16 + lr] = (bf16)p;
            }
        }
#pragma unroll
        for (int e = 0; e < 8; ++e) {
            rsum[e] += __shfl_xor(rsum[e], 1, 32);
            rsum[e] += __shfl_xor(rsum[e], 2, 32);
            rsum[e] += __shfl_xor(rsum[e], 4, 32);
            rsum[e] += __shfl_xor(rsum[e], 8, 32);
            lrow[e] = lrow[e] * alpha[e] + rsum[e];
        }
#pragma unroll
        for (int n = 0; n < 4; ++n)
#pragma unroll
            for (int e = 0; e < 8; ++e) O[n][e] *= alpha[e];

        // wave-internal LDS ordering before reading back P fragments
        __builtin_amdgcn_wave_barrier();
        asm volatile("" ::: "memory");

        // O += P * V
#pragma unroll
        for (int kc = 0; kc < 2; ++kc) {
            v16bf pf = load_frag_a(&Plds[wid][lr][kc * 32 + hi * 16]);
#pragma unroll
            for (int n = 0; n < 4; ++n) {
                v16bf vf = load_frag_b(&Vt[n * 16 + lr][kc * 32 + hi * 16]);
                O[n] = wmma_bf16(pf, vf, O[n]);
            }
        }
        // next iteration's top-of-loop barrier separates these reads from
        // the Vt/Klds overwrites
    }

    // normalize and store bf16 attention output for the projection GEMM
#pragma unroll
    for (int n = 0; n < 4; ++n) {
#pragma unroll
        for (int e = 0; e < 8; ++e) {
            const int qt = qrow0 + hi * 8 + e;
            attn_ws[((size_t)(b * Tc + qt)) * Dc + head * Hc + n * 16 + lr] =
                (bf16)(O[n][e] / lrow[e]);
        }
    }
}

// ---------------------------------------------------------------------------
extern "C" void kernel_launch(void* const* d_in, const int* in_sizes, int n_in,
                              void* d_out, int out_size, void* d_ws, size_t ws_size,
                              hipStream_t stream)
{
    const float* x      = (const float*)d_in[0];
    const float* w_attn = (const float*)d_in[1];
    const float* b_attn = (const float*)d_in[2];
    const float* w_proj = (const float*)d_in[3];
    const float* b_proj = (const float*)d_in[4];

    const size_t M = (size_t)Bc * Tc;            // 4096 rows
    float* out   = (float*)d_out;                // (B,T,D)
    float* k_out = out + M * Dc;                 // (B,T,N,H) == contiguous slice
    float* v_out = out + 2 * M * Dc;

    // workspace: bf16 shadows of q, k, v and attention output (4 x 8 MB)
    bf16* q_ws = (bf16*)d_ws;
    bf16* k_ws = q_ws + M * Dc;
    bf16* v_ws = k_ws + M * Dc;
    bf16* a_ws = v_ws + M * Dc;

    dim3 blk(256);
    // 1) qkv = x @ w_attn + b_attn   (M=4096, N=3072, K=1024)
    gemm_bf16_wmma<false, true><<<dim3(32, 24), blk, 0, stream>>>(
        x, w_attn, b_attn, nullptr, q_ws, k_ws, v_ws, k_out, v_out, Dc, 3 * Dc);
    // 2) causal flash attention  (B * NHEAD * T/128 = 512 blocks)
    attn_flash_wmma<<<dim3(512), blk, 0, stream>>>(q_ws, k_ws, v_ws, a_ws);
    // 3) output = attn @ w_proj + b_proj  (M=4096, N=1024, K=1024)
    gemm_bf16_wmma<true, false><<<dim3(32, 8), blk, 0, stream>>>(
        a_ws, w_proj, b_proj, out, nullptr, nullptr, nullptr, nullptr, nullptr,
        Dc, Dc);
}
